// LSTM_64914135712352
// MI455X (gfx1250) — matrix operand order repaired and checked by reading, not compiled, not run on hardware
//
#include <hip/hip_runtime.h>
#include <hip/hip_bf16.h>
#include <stdint.h>

typedef __attribute__((ext_vector_type(16))) __bf16 v16bf;
typedef __attribute__((ext_vector_type(8)))  float  v8f;

#define T_STEPS 512
#define BATCH   64
#define HID     1024
#define NL      2
#define GATE4   4096            // 4*HID
#define NTILES  256             // GATE4/16
#define KTILES  64              // 2*HID/32
#define GRID_P  64
#define BLK_P   256

// ---------------- workspace layout (bytes) ----------------
// [0, 33554432)            : Wt  bf16, 2 layers x 2048 x 4096, WMMA-B swizzled
// [33554432, 34078720)     : hS  f32  [L][B][H]
// [34078720, 34603008)     : cS  f32  [L][B][H]
// [34603008, 35651584)     : ifgo f32 [B][4H]
// [35651584, +8)           : barrier counter + generation

__device__ inline float sigmoidf_(float x) { return 1.0f / (1.0f + __expf(-x)); }
__device__ inline float tanhf_(float x) {
    float e = __expf(-2.0f * fabsf(x));
    float t = (1.0f - e) / (1.0f + e);
    return copysignf(t, x);
}

__device__ inline void grid_barrier(unsigned* cnt, unsigned* gen, unsigned nblk) {
    __syncthreads();
    if (threadIdx.x == 0) {
        __threadfence();
        unsigned g = __hip_atomic_load(gen, __ATOMIC_ACQUIRE, __HIP_MEMORY_SCOPE_AGENT);
        unsigned a = __hip_atomic_fetch_add(cnt, 1u, __ATOMIC_ACQ_REL, __HIP_MEMORY_SCOPE_AGENT);
        if (a == nblk - 1u) {
            __hip_atomic_store(cnt, 0u, __ATOMIC_RELAXED, __HIP_MEMORY_SCOPE_AGENT);
            __hip_atomic_store(gen, g + 1u, __ATOMIC_RELEASE, __HIP_MEMORY_SCOPE_AGENT);
        } else {
            while (__hip_atomic_load(gen, __ATOMIC_ACQUIRE, __HIP_MEMORY_SCOPE_AGENT) == g)
                __builtin_amdgcn_s_sleep(2);
        }
    }
    __syncthreads();
}

// sum-pair reduction over a 256-thread block (8 wave32)
__device__ inline void block_reduce2(float& a, float& b, float* lds, int tid) {
    #pragma unroll
    for (int off = 16; off; off >>= 1) {
        a += __shfl_xor(a, off, 32);
        b += __shfl_xor(b, off, 32);
    }
    __syncthreads();
    if ((tid & 31) == 0) { lds[tid >> 5] = a; lds[8 + (tid >> 5)] = b; }
    __syncthreads();
    a = 0.f; b = 0.f;
    #pragma unroll
    for (int i = 0; i < 8; ++i) { a += lds[i]; b += lds[8 + i]; }
}

// ------------------------------------------------------------------
// Convert f32 [Wx;Wh] into bf16 tiles laid out exactly as the per-lane
// WMMA B-matrix fragment in this kernel's k order, same as A (16x16x32 bf16): for a 32x16 K-chunk, lane L
// holds N = L%16, K = 32*kt + 16*(L/16) + j  (j = 0..15, contiguous).
// Tile order: [l][ntile][ktile], 512 bf16 per tile.
// ------------------------------------------------------------------
__global__ void convert_weights(const float* __restrict__ Wh,
                                const float* __restrict__ Wx,
                                __bf16* __restrict__ Wt) {
    int gid  = blockIdx.x * blockDim.x + threadIdx.x;
    int wid  = gid >> 5;          // tile id = (l*256 + nt)*64 + kt
    int lane = gid & 31;
    int kt = wid & 63;
    int nt = (wid >> 6) & 255;
    int l  = wid >> 14;
    if (l >= NL) return;
    int n     = nt * 16 + (lane & 15);
    int kbase = kt * 32 + 16 * (lane >> 4);
    __bf16* dst = Wt + (size_t)wid * 512 + lane * 16;
    #pragma unroll
    for (int j = 0; j < 16; ++j) {
        int k = kbase + j;
        float v = (k < HID) ? Wx[((size_t)l * HID + k) * GATE4 + n]
                            : Wh[((size_t)l * HID + (k - HID)) * GATE4 + n];
        dst[j] = (__bf16)v;
    }
}

__global__ void init_state(const float* __restrict__ h0, const float* __restrict__ c0,
                           float* __restrict__ hS, float* __restrict__ cS,
                           unsigned* __restrict__ bar) {
    int i = blockIdx.x * blockDim.x + threadIdx.x;
    if (i < NL * BATCH * HID) { hS[i] = h0[i]; cS[i] = c0[i]; }
    if (i == 0) { bar[0] = 0u; bar[1] = 0u; }
}

__global__ void copy_final(const float* __restrict__ hS, const float* __restrict__ cS,
                           float* __restrict__ dst) {
    int i = blockIdx.x * blockDim.x + threadIdx.x;
    const int n = NL * BATCH * HID;
    if (i < n) { dst[i] = hS[i]; dst[n + i] = cS[i]; }
}

// ------------------------------------------------------------------
// Persistent LayerNorm-LSTM: 64 blocks x 256 threads (512 wave32).
// Phase A: cooperative A-fragment staging into LDS (each wave converts
//          8 of 64 K-tiles), then each wave computes TWO 16x16 tiles of
//          ifgo = [x|h] @ [Wx;Wh] — per K-step: 1 LDS A read (ds_load_b128)
//          + 2 streamed B fragments + 2 v_wmma_f32_16x16x32_bf16 on
//          independent accumulators.
// Phase B: block b handles batch row b (5x layernorm + gate math).
// Agent-scope grid barrier between phases.
// ------------------------------------------------------------------
__global__ void __launch_bounds__(BLK_P, 1)
lstm_persistent(const float* __restrict__ x,
                const float* __restrict__ bh, const float* __restrict__ bx,
                const float* __restrict__ ln_g, const float* __restrict__ ln_b,
                const __bf16* __restrict__ Wt,
                float* __restrict__ hS, float* __restrict__ cS,
                float* __restrict__ ifgo, float* __restrict__ out,
                unsigned* __restrict__ bar) {
    const int wg   = blockIdx.x;            // 0..63
    const int tid  = threadIdx.x;
    const int wave = tid >> 5;              // 0..7
    const int lane = tid & 31;
    const int half = lane >> 4;             // 0: lo half, 1: hi half
    const int m_tile  = wg >> 4;            // 0..3 (batch rows /16)
    const int n_pair  = ((wg & 15) << 3) + wave;   // 0..127
    const int n_tile0 = n_pair * 2;                // 0..254 (even)
    const int mrow  = m_tile * 16 + (lane & 15);
    const int ncol0 = n_tile0 * 16 + (lane & 15);

    __shared__ __align__(16) __bf16 ldsA[KTILES * 512];  // 64 KB A fragments
    __shared__ float red[16];

    #pragma unroll 1
    for (int t = 0; t < T_STEPS; ++t) {
        #pragma unroll 1
        for (int l = 0; l < NL; ++l) {
            // ---------------- Phase A: fused GEMM ----------------
            const float* inp  = (l == 0) ? (x + (size_t)t * BATCH * HID) : hS;
            const float* hrec = hS + (size_t)l * BATCH * HID;
            const __bf16* wt0 = Wt + ((size_t)(l * NTILES + n_tile0) * KTILES) * 512;
            const __bf16* wt1 = wt0 + (size_t)KTILES * 512;

            // Cooperative A staging: wave w converts K-tiles w, w+8, ...
            #pragma unroll 1
            for (int kt = wave; kt < KTILES; kt += 8) {
                const float* src = (kt < 32)
                    ? (inp  + (size_t)mrow * HID + kt * 32)
                    : (hrec + (size_t)mrow * HID + (kt - 32) * 32);
                // 16-bit A layout: lane holds K = {16*half..+16} (contiguous)
                const float* p0 = src + 16 * half;
                const float* p1 = src + 16 * half + 8;
                v16bf a;
                #pragma unroll
                for (int j = 0; j < 8; ++j) a[j]     = (__bf16)p0[j];
                #pragma unroll
                for (int j = 0; j < 8; ++j) a[8 + j] = (__bf16)p1[j];
                *(v16bf*)(ldsA + (size_t)kt * 512 + lane * 16) = a;
            }
            __syncthreads();

            v8f acc0 = {};
            v8f acc1 = {};
            #pragma unroll 1
            for (int kt = 0; kt < KTILES; ++kt) {
                v16bf a  = *(const v16bf*)(ldsA + (size_t)kt * 512 + lane * 16);
                v16bf b0 = *(const v16bf*)(wt0 + (size_t)kt * 512 + lane * 16);
                v16bf b1 = *(const v16bf*)(wt1 + (size_t)kt * 512 + lane * 16);
                if (kt + 1 < KTILES) {
                    __builtin_prefetch(wt0 + (size_t)(kt + 1) * 512 + lane * 16, 0, 3);
                    __builtin_prefetch(wt1 + (size_t)(kt + 1) * 512 + lane * 16, 0, 3);
                }
                acc0 = __builtin_amdgcn_wmma_f32_16x16x32_bf16(
                         false, a, false, b0, (short)0, acc0, false, false);
                acc1 = __builtin_amdgcn_wmma_f32_16x16x32_bf16(
                         false, a, false, b1, (short)0, acc1, false, false);
            }
            {   // C layout: VGPR v -> M = v + 8*half, N = lane%16
                float* od0 = ifgo + ncol0;
                float* od1 = od0 + 16;
                #pragma unroll
                for (int v = 0; v < 8; ++v) {
                    int m = m_tile * 16 + v + 8 * half;
                    od0[(size_t)m * GATE4] = acc0[v];
                    od1[(size_t)m * GATE4] = acc1[v];
                }
            }
            grid_barrier(bar, bar + 1, GRID_P);

            // ---------------- Phase B: LN + gates ----------------
            {
                const int b  = wg;                    // one batch row per block
                const int i0 = tid * 4;               // 256 thr * 4 = H
                float gv[4][4];
                #pragma unroll
                for (int g = 0; g < 4; ++g) {
                    float s = 0.f, s2 = 0.f;
                    #pragma unroll
                    for (int j = 0; j < 4; ++j) {
                        size_t o = (size_t)g * HID + i0 + j;
                        float v = ifgo[(size_t)b * GATE4 + o]
                                + bx[(size_t)l * GATE4 + o]
                                + bh[(size_t)l * GATE4 + o];
                        gv[g][j] = v;
                        s += v; s2 += v * v;
                    }
                    block_reduce2(s, s2, red, tid);
                    float mu  = s * (1.0f / HID);
                    float var = s2 * (1.0f / HID) - mu * mu;
                    float rs  = rsqrtf(var + 1e-5f);
                    const float* gG = ln_g + ((size_t)l * 5 + g) * HID + i0;
                    const float* gB = ln_b + ((size_t)l * 5 + g) * HID + i0;
                    #pragma unroll
                    for (int j = 0; j < 4; ++j)
                        gv[g][j] = (gv[g][j] - mu) * rs * gG[j] + gB[j];
                }
                float* cRow = cS + ((size_t)l * BATCH + b) * HID + i0;
                float* hRow = hS + ((size_t)l * BATCH + b) * HID + i0;
                float cn[4];
                float s = 0.f, s2 = 0.f;
                #pragma unroll
                for (int j = 0; j < 4; ++j) {
                    float c = sigmoidf_(gv[1][j]) * cRow[j]
                            + sigmoidf_(gv[0][j]) * tanhf_(gv[2][j]);
                    cn[j] = c; s += c; s2 += c * c;
                }
                block_reduce2(s, s2, red, tid);
                float mu  = s * (1.0f / HID);
                float var = s2 * (1.0f / HID) - mu * mu;
                float rs  = rsqrtf(var + 1e-5f);
                const float* gG = ln_g + ((size_t)l * 5 + 4) * HID + i0;
                const float* gB = ln_b + ((size_t)l * 5 + 4) * HID + i0;
                #pragma unroll
                for (int j = 0; j < 4; ++j) {
                    float lnc = (cn[j] - mu) * rs * gG[j] + gB[j];
                    float h   = sigmoidf_(gv[3][j]) * tanhf_(lnc);
                    cRow[j] = cn[j];
                    hRow[j] = h;
                    if (l == NL - 1)
                        out[((size_t)t * BATCH + b) * HID + i0 + j] = h;
                }
            }
            grid_barrier(bar, bar + 1, GRID_P);
        }
    }
}

extern "C" void kernel_launch(void* const* d_in, const int* in_sizes, int n_in,
                              void* d_out, int out_size, void* d_ws, size_t ws_size,
                              hipStream_t stream) {
    const float* x   = (const float*)d_in[0];
    const float* h0  = (const float*)d_in[1];
    const float* c0  = (const float*)d_in[2];
    const float* Wh  = (const float*)d_in[3];
    const float* bh  = (const float*)d_in[4];
    const float* Wx  = (const float*)d_in[5];
    const float* bx  = (const float*)d_in[6];
    const float* lng = (const float*)d_in[7];
    const float* lnb = (const float*)d_in[8];
    float* out = (float*)d_out;

    char* ws = (char*)d_ws;
    __bf16*   Wt   = (__bf16*)ws;
    float*    hS   = (float*)(ws + 33554432);
    float*    cS   = (float*)(ws + 34078720);
    float*    ifgo = (float*)(ws + 34603008);
    unsigned* bar  = (unsigned*)(ws + 35651584);

    convert_weights<<<4096, 256, 0, stream>>>(Wh, Wx, Wt);
    init_state<<<512, 256, 0, stream>>>(h0, c0, hS, cS, bar);
    lstm_persistent<<<GRID_P, BLK_P, 0, stream>>>(x, bh, bx, lng, lnb,
                                                  Wt, hS, cS, ifgo, out, bar);
    copy_final<<<512, 256, 0, stream>>>(hS, cS, out + (size_t)T_STEPS * BATCH * HID);
}
